// LinearMultiHeadAttention_9732395893253
// MI455X (gfx1250) — matrix operand order repaired and checked by reading, not compiled, not run on hardware
//
#include <hip/hip_runtime.h>
#include <hip/hip_bf16.h>
#include <stdint.h>

#define B_   4
#define S_   4096
#define DM   1024
#define H_   16
#define HD_  64
#define EPS_ 1e-6f

#define USE_ASYNC_LDS 1

typedef __attribute__((ext_vector_type(16))) __bf16        v16bf;
typedef __attribute__((ext_vector_type(8)))  float         v8f;
typedef __attribute__((ext_vector_type(4)))  unsigned int  u32x4;
typedef __attribute__((ext_vector_type(2)))  unsigned int  u32x2;
typedef __attribute__((ext_vector_type(4)))  float         f32x4;

union Frag16 {
    v16bf bf;
    u32x4 q[2];
};

__device__ __forceinline__ unsigned short f2bf(float f) {
    union { float f; uint32_t u; } v; v.f = f;
    uint32_t u = v.u;
    uint32_t r = u + 0x7FFFu + ((u >> 16) & 1u);   // round-to-nearest-even
    return (unsigned short)(r >> 16);
}
__device__ __forceinline__ float bf2f(unsigned short h) {
    union { float f; uint32_t u; } v; v.u = ((uint32_t)h) << 16;
    return v.f;
}
__device__ __forceinline__ unsigned pack2bf(float lo, float hi) {
    return (unsigned)f2bf(lo) | ((unsigned)f2bf(hi) << 16);
}
__device__ __forceinline__ v8f vzero8() {
    v8f z = {0.f, 0.f, 0.f, 0.f, 0.f, 0.f, 0.f, 0.f};
    return z;
}

// ---- gfx1250 async global->LDS copy (ASYNCcnt path) ----------------------
// LDS byte offset = low 32 bits of the generic pointer (LDS aperture keeps
// the offset in addr[31:0]).
__device__ __forceinline__ void async_ld16(void* lds, const void* g) {
#if USE_ASYNC_LDS
    unsigned lds_off = (unsigned)(size_t)lds;
    unsigned long long ga = (unsigned long long)(size_t)g;
    asm volatile("global_load_async_to_lds_b128 %0, %1, off"
                 :: "v"(lds_off), "v"(ga) : "memory");
#else
    *(u32x4*)lds = *(const u32x4*)g;
#endif
}
__device__ __forceinline__ void async_wait0() {
#if USE_ASYNC_LDS
    asm volatile("s_wait_asynccnt 0x0" ::: "memory");
#endif
}

// ---------------------------------------------------------------------------
// Y[m][n] = sum_k A[m][k] * W[n][k] + bias[n]      (M=16384, N=K=1024)
// Block tile 128x128, K-step 32, double-buffered LDS (one barrier / K-step).
// 8 waves: 4 (M) x 2 (N); each wave 2x4 WMMA tiles.
// ---------------------------------------------------------------------------
template<bool A_BF16, bool OUT_BF16>
__global__ __launch_bounds__(256)
void gemm_xt_bias(const void* __restrict__ Aptr, const float* __restrict__ W,
                  const float* __restrict__ bias, void* __restrict__ Yptr) {
    const int K = 1024, N = 1024;
    __shared__ unsigned short As[2][128 * 48];   // [m][k], stride 48 (96B)
    __shared__ unsigned short Bs[2][128 * 48];   // [n][k]

    const int tid  = threadIdx.x;
    const int wid  = tid >> 5;
    const int lane = tid & 31;
    const int lrow = lane & 15;
    const int lhi  = lane >> 4;
    const int m0 = blockIdx.y * 128;
    const int n0 = blockIdx.x * 128;
    const int wm = wid >> 1;     // 0..3 -> 32 M-rows
    const int wn = wid & 1;      // 0..1 -> 64 N-cols

    const float*          Af = (const float*)Aptr;
    const unsigned short* Ab = (const unsigned short*)Aptr;

    auto issueA_async = [&](int buf, int k0) {
#pragma unroll
        for (int it = 0; it < 2; ++it) {
            int slot = tid + it * 256;               // 512 = 128 rows x 4 x 16B
            int row = slot >> 2, c8 = slot & 3;
            async_ld16(&As[buf][row * 48 + c8 * 8],
                       Ab + (size_t)(m0 + row) * K + k0 + c8 * 8);
        }
    };
    auto loadA_regs = [&](int k0, f32x4* r) {
#pragma unroll
        for (int it = 0; it < 4; ++it) {
            int slot = tid + it * 256;               // 1024 = 128 rows x 8 f32x4
            int row = slot >> 3, c4 = slot & 7;
            r[it] = *(const f32x4*)(Af + (size_t)(m0 + row) * K + k0 + c4 * 4);
        }
    };
    auto storeA_lds = [&](int buf, const f32x4* r) {
#pragma unroll
        for (int it = 0; it < 4; ++it) {
            int slot = tid + it * 256;
            int row = slot >> 3, c4 = slot & 7;
            u32x2 pk = {pack2bf(r[it].x, r[it].y), pack2bf(r[it].z, r[it].w)};
            *(u32x2*)(&As[buf][row * 48 + c4 * 4]) = pk;
        }
    };
    auto loadB_regs = [&](int k0, f32x4* r) {
#pragma unroll
        for (int it = 0; it < 4; ++it) {
            int slot = tid + it * 256;
            int row = slot >> 3, c4 = slot & 7;
            r[it] = *(const f32x4*)(W + (size_t)(n0 + row) * K + k0 + c4 * 4);
        }
    };
    auto storeB_lds = [&](int buf, const f32x4* r) {
#pragma unroll
        for (int it = 0; it < 4; ++it) {
            int slot = tid + it * 256;
            int row = slot >> 3, c4 = slot & 7;
            u32x2 pk = {pack2bf(r[it].x, r[it].y), pack2bf(r[it].z, r[it].w)};
            *(u32x2*)(&Bs[buf][row * 48 + c4 * 4]) = pk;
        }
    };

    v8f c[2][4];
#pragma unroll
    for (int i = 0; i < 2; ++i)
#pragma unroll
        for (int j = 0; j < 4; ++j) c[i][j] = vzero8();

    f32x4 ra[4], rb[4];

    // ---- prologue: stage k0 = 0 into buffer 0 ----
    if (A_BF16) {
        issueA_async(0, 0);
    } else {
        loadA_regs(0, ra);
        storeA_lds(0, ra);
    }
    loadB_regs(0, rb);
    storeB_lds(0, rb);
    if (A_BF16) async_wait0();

    int cur = 0;
    for (int k0 = 0; k0 < K; k0 += 32) {
        const int  nxt  = cur ^ 1;
        const bool more = (k0 + 32) < K;

        // prefetch next tile into registers (f32 paths) before the barrier
        if (!A_BF16 && more) loadA_regs(k0 + 32, ra);
        if (more)            loadB_regs(k0 + 32, rb);

        __syncthreads();     // buf[cur] visible to all; buf[nxt] free to write

        // async A copy for next tile overlaps the WMMAs below
        if (A_BF16 && more) issueA_async(nxt, k0 + 32);

        // ---- fragments from buf[cur] ----
        Frag16 a[2], b[4];
        const int kb = lhi * 8;     // A: lanes<16 -> K 0..7/16..23 ; >=16 -> 8..15/24..31
        const int kr = lhi * 8;     // B: lanes<16 -> K 0..7/16..23 ; >=16 -> 8..15/24..31
#pragma unroll
        for (int i = 0; i < 2; ++i) {
            int row = wm * 32 + i * 16 + lrow;
            a[i].q[0] = *(const u32x4*)(&As[cur][row * 48 + kb]);
            a[i].q[1] = *(const u32x4*)(&As[cur][row * 48 + kb + 16]);
        }
#pragma unroll
        for (int j = 0; j < 4; ++j) {
            int row = wn * 64 + j * 16 + lrow;
            b[j].q[0] = *(const u32x4*)(&Bs[cur][row * 48 + kr]);
            b[j].q[1] = *(const u32x4*)(&Bs[cur][row * 48 + kr + 16]);
        }
#pragma unroll
        for (int i = 0; i < 2; ++i)
#pragma unroll
            for (int j = 0; j < 4; ++j)
                c[i][j] = __builtin_amdgcn_wmma_f32_16x16x32_bf16(
                    false, a[i].bf, false, b[j].bf, (short)0, c[i][j], false, false);

        // stage next tile into buf[nxt] (convert paths), finish async copy
        if (more) {
            if (!A_BF16) storeA_lds(nxt, ra);
            storeB_lds(nxt, rb);
            if (A_BF16) async_wait0();
        }
        cur = nxt;
    }

    // ---- epilogue: C/D layout (VGPR r: M = r + 8*lhi, N = lrow) ----
#pragma unroll
    for (int i = 0; i < 2; ++i) {
#pragma unroll
        for (int j = 0; j < 4; ++j) {
            int nn = n0 + wn * 64 + j * 16 + lrow;
            float bv = bias[nn];
#pragma unroll
            for (int r = 0; r < 8; ++r) {
                int mm = m0 + wm * 32 + i * 16 + r + lhi * 8;
                float val = c[i][j][r] + bv;
                if (OUT_BF16)
                    ((unsigned short*)Yptr)[(size_t)mm * N + nn] = f2bf(val);
                else
                    ((float*)Yptr)[(size_t)mm * N + nn] = val;
            }
        }
    }
}

// ---------------------------------------------------------------------------
// Causal depthwise conv (KS=3) + optional feature map elu(x)+1.
// 8 channels per thread (vectorized bf16 traffic).
// ---------------------------------------------------------------------------
__global__ __launch_bounds__(256)
void conv_fmap(const unsigned short* __restrict__ P, const float* __restrict__ cw,
               const float* __restrict__ cb, unsigned short* __restrict__ out, int fmap) {
    size_t vid = (size_t)blockIdx.x * 256 + threadIdx.x;
    size_t idx = vid * 8;
    int d0  = (int)(idx & (DM - 1));
    size_t bs = idx >> 10;                 // b*S + s
    int s   = (int)(bs & (S_ - 1));
    int dk0 = d0 & (HD_ - 1);

    u32x4 zz = {0u, 0u, 0u, 0u};
    u32x4 x0 = *(const u32x4*)(P + idx);
    u32x4 x1 = (s >= 1) ? *(const u32x4*)(P + idx - DM)     : zz;
    u32x4 x2 = (s >= 2) ? *(const u32x4*)(P + idx - 2 * DM) : zz;
    const unsigned short* p0 = (const unsigned short*)&x0;
    const unsigned short* p1 = (const unsigned short*)&x1;
    const unsigned short* p2 = (const unsigned short*)&x2;

    unsigned short res[8];
#pragma unroll
    for (int i = 0; i < 8; ++i) {
        int dk = dk0 + i;
        float y = cb[dk] + cw[dk * 3 + 0] * bf2f(p2[i])
                         + cw[dk * 3 + 1] * bf2f(p1[i])
                         + cw[dk * 3 + 2] * bf2f(p0[i]);
        if (fmap) y = (y > 0.0f) ? (y + 1.0f) : __expf(y);   // elu(y)+1
        res[i] = f2bf(y);
    }
    u32x4 o;
    const unsigned* rp = (const unsigned*)res;
    o.x = rp[0]; o.y = rp[1]; o.z = rp[2]; o.w = rp[3];
    *(u32x4*)(out + idx) = o;
}

// ---------------------------------------------------------------------------
// Ksum[b, d] = sum_s kf[b, s, d]    (d = h*64 + dk); one block per (b,d)
// ---------------------------------------------------------------------------
__global__ __launch_bounds__(256)
void ksum_kernel(const unsigned short* __restrict__ kf, float* __restrict__ ksum) {
    int bd = blockIdx.x;
    int b = bd >> 10;
    int d = bd & (DM - 1);
    __shared__ float red[256];
    float acc = 0.f;
    for (int s = threadIdx.x; s < S_; s += 256)
        acc += bf2f(kf[((size_t)(b * S_ + s) << 10) + d]);
    red[threadIdx.x] = acc;
    __syncthreads();
    for (int off = 128; off > 0; off >>= 1) {
        if (threadIdx.x < off) red[threadIdx.x] += red[threadIdx.x + off];
        __syncthreads();
    }
    if (threadIdx.x == 0) ksum[bd] = red[0];
}

__global__ __launch_bounds__(256)
void zero_f32(float* __restrict__ p) {
    p[(size_t)blockIdx.x * 256 + threadIdx.x] = 0.0f;
}

// ---------------------------------------------------------------------------
// Partial KV^T: grid (8 s-chunks, B*H). Accumulates 512 s per block via WMMA
// (64 s staged per iteration, transposed into LDS with packed b32 stores),
// then global_atomic_add_f32 into kvf[b,h,e,d].
// ---------------------------------------------------------------------------
__global__ __launch_bounds__(256)
void kv_partial(const unsigned short* __restrict__ kf, const unsigned short* __restrict__ vf,
                float* __restrict__ kvf) {
    int bh = blockIdx.y;
    int bb = bh >> 4, h = bh & 15;
    int sbeg = blockIdx.x * (S_ / 8);        // 512 s per block
    __shared__ unsigned short kT[64 * 72];   // [dk][s] , 64 s staged, stride 72 (144B)
    __shared__ unsigned short vT[64 * 72];   // [e][s]
    int tid = threadIdx.x, wid = tid >> 5, lane = tid & 31;
    int lrow = lane & 15, lhi = lane >> 4;
    int t0 = wid * 2;                        // wave owns tiles t0, t0+1 (same d-row)
    int d0 = (t0 >> 2) * 16;
    int e0 = (t0 & 3) * 16;
    v8f c0 = vzero8(), c1 = vzero8();
    int g = tid & 7, spair = tid >> 3;       // 32 s-pairs x 8 channel-groups

    for (int s0 = sbeg; s0 < sbeg + S_ / 8; s0 += 64) {
        size_t base = ((size_t)(bb * S_ + s0 + spair * 2) << 10) + h * 64 + g * 8;
        u32x4 ka = *(const u32x4*)(kf + base);
        u32x4 kc = *(const u32x4*)(kf + base + DM);
        u32x4 va = *(const u32x4*)(vf + base);
        u32x4 vc = *(const u32x4*)(vf + base + DM);
        const unsigned short* ka_ = (const unsigned short*)&ka;
        const unsigned short* kc_ = (const unsigned short*)&kc;
        const unsigned short* va_ = (const unsigned short*)&va;
        const unsigned short* vc_ = (const unsigned short*)&vc;
        __syncthreads();
#pragma unroll
        for (int i = 0; i < 8; ++i) {
            // pack (s even, s odd) -> one b32 store
            *(unsigned*)(&kT[(g * 8 + i) * 72 + spair * 2]) =
                (unsigned)ka_[i] | ((unsigned)kc_[i] << 16);
            *(unsigned*)(&vT[(g * 8 + i) * 72 + spair * 2]) =
                (unsigned)va_[i] | ((unsigned)vc_[i] << 16);
        }
        __syncthreads();
        const int kb = lhi * 8, kr = lhi * 8;
#pragma unroll
        for (int ks = 0; ks < 64; ks += 32) {
            Frag16 a, b0, b1;
            a.q[0]  = *(const u32x4*)(&kT[(d0 + lrow) * 72 + ks + kb]);
            a.q[1]  = *(const u32x4*)(&kT[(d0 + lrow) * 72 + ks + kb + 16]);
            b0.q[0] = *(const u32x4*)(&vT[(e0 + lrow) * 72 + ks + kr]);
            b0.q[1] = *(const u32x4*)(&vT[(e0 + lrow) * 72 + ks + kr + 16]);
            b1.q[0] = *(const u32x4*)(&vT[(e0 + 16 + lrow) * 72 + ks + kr]);
            b1.q[1] = *(const u32x4*)(&vT[(e0 + 16 + lrow) * 72 + ks + kr + 16]);
            c0 = __builtin_amdgcn_wmma_f32_16x16x32_bf16(false, a.bf, false, b0.bf, (short)0, c0, false, false);
            c1 = __builtin_amdgcn_wmma_f32_16x16x32_bf16(false, a.bf, false, b1.bf, (short)0, c1, false, false);
        }
    }

    size_t ob = (size_t)bh * 64 * 64;
#pragma unroll
    for (int r = 0; r < 8; ++r) {
        int dd = d0 + r + lhi * 8;
        int ea = e0 + lrow;
        atomicAdd(&kvf[ob + (size_t)ea * 64 + dd], c0[r]);
        atomicAdd(&kvf[ob + (size_t)(ea + 16) * 64 + dd], c1[r]);
    }
}

__global__ __launch_bounds__(256)
void kv_convert(const float* __restrict__ kvf, unsigned short* __restrict__ kvT) {
    size_t i = (size_t)blockIdx.x * 256 + threadIdx.x;
    kvT[i] = f2bf(kvf[i]);
}

// ---------------------------------------------------------------------------
// V_new[b,s,h,e] = Z[s] * sum_d qf[b,s,h,d] * KV[d][e],  Z = 1/(q . (Ksum+eps))
// Grid: (S/128, B*H). 8 waves, each 16 M-rows x 4 N-tiles; K=64 in 2 steps.
// KV^T staged into LDS via async global->LDS copies.
// ---------------------------------------------------------------------------
__global__ __launch_bounds__(256)
void vnew_kernel(const unsigned short* __restrict__ qf, const unsigned short* __restrict__ kvT,
                 const float* __restrict__ ksum, unsigned short* __restrict__ vnew) {
    int bh = blockIdx.y;
    int bb = bh >> 4, h = bh & 15;
    int s0 = blockIdx.x * 128;
    __shared__ unsigned short kvs[64 * 64];   // [e][d]
    __shared__ float kse[64];
    __shared__ float zbuf[128];
    int tid = threadIdx.x, wid = tid >> 5, lane = tid & 31;
    int lrow = lane & 15, lhi = lane >> 4;

    {
        const unsigned short* g = kvT + (size_t)bh * 4096;
        async_ld16(&kvs[tid * 8],        g + tid * 8);
        async_ld16(&kvs[tid * 8 + 2048], g + tid * 8 + 2048);
    }
    if (tid < 64) kse[tid] = ksum[bb * DM + h * 64 + tid] + EPS_;
    async_wait0();
    __syncthreads();

    if (tid < 128) {
        int s = s0 + tid;
        const unsigned short* qrow = qf + ((size_t)(bb * S_ + s) << 10) + h * 64;
        float dot = 0.f;
#pragma unroll
        for (int d = 0; d < 64; ++d) dot += bf2f(qrow[d]) * kse[d];
        zbuf[tid] = 1.0f / dot;
    }

    v8f c[4];
#pragma unroll
    for (int j = 0; j < 4; ++j) c[j] = vzero8();

    int m = s0 + wid * 16 + lrow;
    const unsigned short* arow = qf + ((size_t)(bb * S_ + m) << 10) + h * 64;
    const int kb = lhi * 8, kr = lhi * 8;
#pragma unroll
    for (int ks = 0; ks < 64; ks += 32) {
        Frag16 a;
        a.q[0] = *(const u32x4*)(arow + ks + kb);
        a.q[1] = *(const u32x4*)(arow + ks + kb + 16);
#pragma unroll
        for (int j = 0; j < 4; ++j) {
            Frag16 bfr;
            int e = j * 16 + lrow;
            bfr.q[0] = *(const u32x4*)(&kvs[e * 64 + ks + kr]);
            bfr.q[1] = *(const u32x4*)(&kvs[e * 64 + ks + kr + 16]);
            c[j] = __builtin_amdgcn_wmma_f32_16x16x32_bf16(
                false, a.bf, false, bfr.bf, (short)0, c[j], false, false);
        }
    }
    __syncthreads();   // zbuf ready

#pragma unroll
    for (int j = 0; j < 4; ++j) {
        int e = j * 16 + lrow;
#pragma unroll
        for (int r = 0; r < 8; ++r) {
            int lr = wid * 16 + r + lhi * 8;
            float val = c[j][r] * zbuf[lr];
            int s = s0 + lr;
            vnew[((size_t)(bb * S_ + s) << 10) + h * 64 + e] = f2bf(val);
        }
    }
}

// ---------------------------------------------------------------------------
extern "C" void kernel_launch(void* const* d_in, const int* in_sizes, int n_in,
                              void* d_out, int out_size, void* d_ws, size_t ws_size,
                              hipStream_t stream) {
    const float* query = (const float*)d_in[0];
    const float* key   = (const float*)d_in[1];
    const float* value = (const float*)d_in[2];
    const float* q_w = (const float*)d_in[3];
    const float* q_b = (const float*)d_in[4];
    const float* k_w = (const float*)d_in[5];
    const float* k_b = (const float*)d_in[6];
    const float* v_w = (const float*)d_in[7];
    const float* v_b = (const float*)d_in[8];
    const float* o_w = (const float*)d_in[9];
    const float* o_b = (const float*)d_in[10];
    const float* qc_w = (const float*)d_in[11];
    const float* qc_b = (const float*)d_in[12];
    const float* kc_w = (const float*)d_in[13];
    const float* kc_b = (const float*)d_in[14];
    const float* vc_w = (const float*)d_in[15];
    const float* vc_b = (const float*)d_in[16];

    char* ws = (char*)d_ws;
    const size_t TBF = (size_t)B_ * S_ * DM * sizeof(unsigned short);  // 32MB
    unsigned short* qp  = (unsigned short*)(ws);
    unsigned short* kp  = (unsigned short*)(ws + 1 * TBF);
    unsigned short* vp  = (unsigned short*)(ws + 2 * TBF);
    unsigned short* qfm = (unsigned short*)(ws + 3 * TBF);
    unsigned short* kfm = (unsigned short*)(ws + 4 * TBF);
    unsigned short* vfm = (unsigned short*)(ws + 5 * TBF);
    float*          ksm = (float*)(ws + 6 * TBF);
    unsigned short* kvT = (unsigned short*)(ws + 6 * TBF + (1u << 20));
    float*          kvf = (float*)(ws + 6 * TBF + (2u << 20));
    unsigned short* vnw = qp;   // qp dead after conv of q -> reuse

    dim3 gg(1024 / 128, (B_ * S_) / 128);   // (8, 128)
    gemm_xt_bias<false, true><<<gg, 256, 0, stream>>>((const void*)query, q_w, q_b, (void*)qp);
    gemm_xt_bias<false, true><<<gg, 256, 0, stream>>>((const void*)key,   k_w, k_b, (void*)kp);
    gemm_xt_bias<false, true><<<gg, 256, 0, stream>>>((const void*)value, v_w, v_b, (void*)vp);

    int nconv = (B_ * S_ * DM) / (256 * 8);   // 8 channels per thread
    conv_fmap<<<nconv, 256, 0, stream>>>(qp, qc_w, qc_b, qfm, 1);
    conv_fmap<<<nconv, 256, 0, stream>>>(kp, kc_w, kc_b, kfm, 1);
    conv_fmap<<<nconv, 256, 0, stream>>>(vp, vc_w, vc_b, vfm, 0);

    ksum_kernel<<<B_ * DM, 256, 0, stream>>>(kfm, ksm);

    int nkv = (B_ * H_ * 64 * 64) / 256;      // 1024
    zero_f32<<<nkv, 256, 0, stream>>>(kvf);
    kv_partial<<<dim3(8, B_ * H_), 256, 0, stream>>>(kfm, vfm, kvf);
    kv_convert<<<nkv, 256, 0, stream>>>(kvf, kvT);

    vnew_kernel<<<dim3(S_ / 128, B_ * H_), 256, 0, stream>>>(qfm, kvT, ksm, vnw);

    gemm_xt_bias<true, false><<<gg, 256, 0, stream>>>((const void*)vnw, o_w, o_b, d_out);
}
